// GATNet_82351702934258
// MI455X (gfx1250) — matrix operand order repaired and checked by reading, not compiled, not run on hardware
//
#include <hip/hip_runtime.h>
#include <hip/hip_bf16.h>
#include <math.h>

#define N_NODES 20000
#define N_EDGES 320000
#define F_IN    128
#define HID     64
#define NHEAD   6
#define FDIM    (NHEAD * HID)      // 384
#define OUT_DIM 10
#define NGRAPH  64
#define NEG_SLOPE 0.2f
#define EE      (N_EDGES + N_NODES) // edges + self loops

// ordered-uint encoding of -inf: f2ord(-inf) = ~(0xFF800000) = 0x007FFFFF
#define ORD_NEG_INF 0x007FFFFFu

typedef __attribute__((ext_vector_type(16))) __bf16 v16bf;
typedef __attribute__((ext_vector_type(8)))  float  v8f;

// ---------------------------------------------------------------------------
// Ordered-uint encoding of float for atomicMax (monotonic over all finite f)
// ---------------------------------------------------------------------------
__device__ __forceinline__ unsigned f2ord(float f) {
    unsigned u = __float_as_uint(f);
    return (u & 0x80000000u) ? ~u : (u | 0x80000000u);
}
__device__ __forceinline__ float ord2f(unsigned u) {
    return (u & 0x80000000u) ? __uint_as_float(u & 0x7FFFFFFFu)
                             : __uint_as_float(~u);
}

// ---------------------------------------------------------------------------
// Split-bf16 WMMA GEMM: C[M,Ncol] = A[M,K] @ B[K,Ncol], fp32 in/out.
// Block: 256 threads = 8 waves; block tile 32x64; wave tile 16x16; K step 32.
// Each operand split hi/lo bf16; 3 WMMAs per K-step -> fp32-class accuracy.
// Requires M%32==0, K%32==0, Ncol%64==0 (true for all call sites).
// ---------------------------------------------------------------------------
__global__ __launch_bounds__(256) void gemm_wmma_bf16x3(
    const float* __restrict__ A, const float* __restrict__ B,
    float* __restrict__ C, int K, int Ncol)
{
    __shared__ float As[32][33];
    __shared__ float Bs[32][65];

    const int tid  = threadIdx.x;
    const int lane = tid & 31;
    const int wave = tid >> 5;
    const int tm   = wave >> 2;          // 0..1  (row sub-tile)
    const int tn   = wave & 3;           // 0..3  (col sub-tile)
    const int rowBase = blockIdx.x * 32;
    const int colBase = blockIdx.y * 64;

    v8f acc = {};

    const int rA  = lane & 15;               // A row within 16x16 tile
    const int kbA = (lane < 16) ? 0 : 8;     // A K-half offset (ISA 16-bit A layout)
    const int cB  = lane & 15;               // B column within tile
    const int khB = (lane < 16) ? 0 : 8;     // B K offset (ISA 16-bit layout, matches A)

    for (int k0 = 0; k0 < K; k0 += 32) {
        // cooperative loads: A 32x32, B 32x64
        #pragma unroll
        for (int t = 0; t < 4; ++t) {
            int idx = tid + t * 256;
            int r = idx >> 5, c = idx & 31;
            As[r][c] = A[(size_t)(rowBase + r) * K + (k0 + c)];
        }
        #pragma unroll
        for (int t = 0; t < 8; ++t) {
            int idx = tid + t * 256;
            int r = idx >> 6, c = idx & 63;
            Bs[r][c] = B[(size_t)(k0 + r) * Ncol + (colBase + c)];
        }
        __syncthreads();

        v16bf a_hi, a_lo, b_hi, b_lo;
        // A fragment per ISA "16-bit A-Matrix 16x32" layout
        #pragma unroll
        for (int e = 0; e < 16; ++e) {
            int j = e >> 1, rb = e & 1;
            int k = (j < 4) ? (kbA + 2 * j + rb)
                            : (16 + kbA + 2 * (j - 4) + rb);
            float v = As[tm * 16 + rA][k];
            __bf16 h = (__bf16)v;
            a_hi[e] = h;
            a_lo[e] = (__bf16)(v - (float)h);
        }
        // B fragment: lane = column, K halves split at lane 16
        #pragma unroll
        for (int e = 0; e < 16; ++e) {
            float v = Bs[khB + e + (e & 8)][tn * 16 + cB];
            __bf16 h = (__bf16)v;
            b_hi[e] = h;
            b_lo[e] = (__bf16)(v - (float)h);
        }

        acc = __builtin_amdgcn_wmma_f32_16x16x32_bf16(false, a_hi, false, b_lo,
                                                      (short)0, acc, false, false);
        acc = __builtin_amdgcn_wmma_f32_16x16x32_bf16(false, a_lo, false, b_hi,
                                                      (short)0, acc, false, false);
        acc = __builtin_amdgcn_wmma_f32_16x16x32_bf16(false, a_hi, false, b_hi,
                                                      (short)0, acc, false, false);
        __syncthreads();
    }

    // C/D layout: VGPR v -> row = v + 8*(lane>=16); col = lane%16
    const int col    = colBase + tn * 16 + (lane & 15);
    const int rowOff = rowBase + tm * 16 + ((lane >> 4) << 3);
    #pragma unroll
    for (int v = 0; v < 8; ++v)
        C[(size_t)(rowOff + v) * Ncol + col] = acc[v];
}

// ---------------------------------------------------------------------------
// Per-(node, head) attention logits: alpha_s = <h[n,h,:], a_src[h,:]>, etc.
// ---------------------------------------------------------------------------
__global__ void alphas_kernel(const float* __restrict__ h,
                              const float* __restrict__ a_src,
                              const float* __restrict__ a_dst,
                              float* __restrict__ als, float* __restrict__ ald)
{
    int idx = blockIdx.x * blockDim.x + threadIdx.x;
    if (idx >= N_NODES * NHEAD) return;
    int n = idx / NHEAD, hd = idx % NHEAD;
    const float* hp  = h + (size_t)n * FDIM + hd * HID;
    const float* asp = a_src + hd * HID;
    const float* adp = a_dst + hd * HID;
    float s = 0.f, d = 0.f;
    #pragma unroll 8
    for (int c = 0; c < HID; ++c) { s += hp[c] * asp[c]; d += hp[c] * adp[c]; }
    als[idx] = s; ald[idx] = d;
}

// ---------------------------------------------------------------------------
// Edge passes (idx over (E+N)*NHEAD; edge id >= E means self-loop)
// ---------------------------------------------------------------------------
__device__ __forceinline__ void edge_sd(int e, const int* __restrict__ ei,
                                        int& s, int& d)
{
    if (e < N_EDGES) { s = ei[e]; d = ei[N_EDGES + e]; }
    else             { s = d = e - N_EDGES; }
}

__global__ void edge_max_kernel(const int* __restrict__ ei,
                                const float* __restrict__ als,
                                const float* __restrict__ ald,
                                unsigned* __restrict__ m)
{
    int idx = blockIdx.x * blockDim.x + threadIdx.x;
    if (idx >= EE * NHEAD) return;
    int e = idx / NHEAD, hd = idx % NHEAD;
    int s, d; edge_sd(e, ei, s, d);
    float v = als[s * NHEAD + hd] + ald[d * NHEAD + hd];
    v = (v > 0.f) ? v : NEG_SLOPE * v;
    atomicMax(&m[d * NHEAD + hd], f2ord(v));
}

__global__ void edge_exp_kernel(const int* __restrict__ ei,
                                const float* __restrict__ als,
                                const float* __restrict__ ald,
                                const unsigned* __restrict__ m,
                                float* __restrict__ ssum,
                                float* __restrict__ exb)
{
    int idx = blockIdx.x * blockDim.x + threadIdx.x;
    if (idx >= EE * NHEAD) return;
    int e = idx / NHEAD, hd = idx % NHEAD;
    int s, d; edge_sd(e, ei, s, d);
    float v = als[s * NHEAD + hd] + ald[d * NHEAD + hd];
    v = (v > 0.f) ? v : NEG_SLOPE * v;
    float ex = __expf(v - ord2f(m[d * NHEAD + hd]));
    exb[idx] = ex;
    atomicAdd(&ssum[d * NHEAD + hd], ex);
}

__global__ void edge_agg_kernel(const int* __restrict__ ei,
                                const float* __restrict__ exb,
                                const float* __restrict__ ssum,
                                const float* __restrict__ hbuf,
                                float* __restrict__ out)
{
    int idx = blockIdx.x * blockDim.x + threadIdx.x;
    if (idx >= EE * NHEAD) return;
    int e = idx / NHEAD, hd = idx % NHEAD;
    int s, d; edge_sd(e, ei, s, d);
    float alpha = exb[idx] / (ssum[d * NHEAD + hd] + 1e-16f);
    const float* hs = hbuf + (size_t)s * FDIM + hd * HID;
    float*       od = out  + (size_t)d * FDIM + hd * HID;
    #pragma unroll 8
    for (int c = 0; c < HID; ++c)
        atomicAdd(&od[c], alpha * hs[c]);
}

__global__ void bias_relu_kernel(float* __restrict__ y, const float* __restrict__ b)
{
    int idx = blockIdx.x * blockDim.x + threadIdx.x;
    if (idx >= N_NODES * FDIM) return;
    float v = y[idx] + b[idx % FDIM];
    y[idx] = (v > 0.f) ? v : 0.f;
}

// ---------------------------------------------------------------------------
// Pool + readout
// ---------------------------------------------------------------------------
__global__ void pool_kernel(const float* __restrict__ P, const float* __restrict__ Q,
                            const int* __restrict__ batch,
                            float* __restrict__ psum, float* __restrict__ cnt)
{
    int idx = blockIdx.x * blockDim.x + threadIdx.x;
    if (idx >= N_NODES * FDIM) return;
    int n = idx / FDIM, f = idx % FDIM;
    int g = batch[n];
    atomicAdd(&psum[(size_t)g * FDIM + f], P[idx] + Q[idx]);
    if (f == 0) atomicAdd(&cnt[g], 1.0f);
}

__global__ __launch_bounds__(FDIM) void readout_kernel(
    const float* __restrict__ psum, const float* __restrict__ cnt,
    const float* __restrict__ r1w, const float* __restrict__ r1b,
    const float* __restrict__ r2w, const float* __restrict__ r2b,
    float* __restrict__ out)
{
    __shared__ float xr[FDIM];
    __shared__ float hid[HID];
    int g = blockIdx.x, t = threadIdx.x;
    float c = cnt[g]; c = (c > 1.f) ? c : 1.f;
    float v = psum[(size_t)g * FDIM + t] / c;
    xr[t] = v;
    out[NGRAPH * OUT_DIM + (size_t)g * FDIM + t] = v;   // x_readout (2nd output)
    __syncthreads();
    if (t < HID) {
        float s = r1b[t];
        for (int f = 0; f < FDIM; ++f) s += xr[f] * r1w[f * HID + t];
        hid[t] = (s > 0.f) ? s : 0.f;
    }
    __syncthreads();
    if (t < OUT_DIM) {
        float s = r2b[t];
        #pragma unroll
        for (int j = 0; j < HID; ++j) s += hid[j] * r2w[j * OUT_DIM + t];
        out[(size_t)g * OUT_DIM + t] = s;               // x_global (1st output)
    }
}

// ---------------------------------------------------------------------------
// Init helpers
// ---------------------------------------------------------------------------
__global__ void zero_f_kernel(float* p, int n) {
    int i = blockIdx.x * blockDim.x + threadIdx.x;
    if (i < n) p[i] = 0.f;
}
__global__ void fill_u32_kernel(unsigned* p, unsigned v, int n) {
    int i = blockIdx.x * blockDim.x + threadIdx.x;
    if (i < n) p[i] = v;
}

// ---------------------------------------------------------------------------
// One GAT convolution layer: out = relu( aggregate(softmax-attn, in@W) + b )
// ---------------------------------------------------------------------------
static void run_gat(const float* in, int K, const float* W,
                    const float* asrc, const float* adst, const float* bias,
                    float* out, float* Hbuf, float* als, float* ald,
                    unsigned* m, float* ssum, float* exb,
                    const int* ei, hipStream_t stream)
{
    dim3 gg(N_NODES / 32, FDIM / 64);
    gemm_wmma_bf16x3<<<gg, 256, 0, stream>>>(in, W, Hbuf, K, FDIM);

    const int nh = N_NODES * NHEAD;
    alphas_kernel<<<(nh + 255) / 256, 256, 0, stream>>>(Hbuf, asrc, adst, als, ald);

    fill_u32_kernel<<<(nh + 255) / 256, 256, 0, stream>>>(m, ORD_NEG_INF, nh);
    zero_f_kernel<<<(nh + 255) / 256, 256, 0, stream>>>(ssum, nh);
    const int tot = N_NODES * FDIM;
    zero_f_kernel<<<(tot + 255) / 256, 256, 0, stream>>>(out, tot);

    const int eh = EE * NHEAD;
    edge_max_kernel<<<(eh + 255) / 256, 256, 0, stream>>>(ei, als, ald, m);
    edge_exp_kernel<<<(eh + 255) / 256, 256, 0, stream>>>(ei, als, ald, m, ssum, exb);
    edge_agg_kernel<<<(eh + 255) / 256, 256, 0, stream>>>(ei, exb, ssum, Hbuf, out);

    bias_relu_kernel<<<(tot + 255) / 256, 256, 0, stream>>>(out, bias);
}

extern "C" void kernel_launch(void* const* d_in, const int* in_sizes, int n_in,
                              void* d_out, int out_size, void* d_ws, size_t ws_size,
                              hipStream_t stream)
{
    (void)in_sizes; (void)n_in; (void)out_size; (void)ws_size;
    const float* x     = (const float*)d_in[0];
    const int*   ei    = (const int*)  d_in[1];
    const int*   batch = (const int*)  d_in[2];
    const float* W1  = (const float*)d_in[3],  *a1s = (const float*)d_in[4];
    const float* a1d = (const float*)d_in[5],  *b1  = (const float*)d_in[6];
    const float* W2  = (const float*)d_in[7],  *a2s = (const float*)d_in[8];
    const float* a2d = (const float*)d_in[9],  *b2  = (const float*)d_in[10];
    const float* Ws1 = (const float*)d_in[11], *as1s= (const float*)d_in[12];
    const float* as1d= (const float*)d_in[13], *bs1 = (const float*)d_in[14];
    const float* Ws2 = (const float*)d_in[15], *as2s= (const float*)d_in[16];
    const float* as2d= (const float*)d_in[17], *bs2 = (const float*)d_in[18];
    const float* r1w = (const float*)d_in[19], *r1b = (const float*)d_in[20];
    const float* r2w = (const float*)d_in[21], *r2b = (const float*)d_in[22];
    float* out = (float*)d_out;

    // workspace carve-up (256B aligned)
    char* ws = (char*)d_ws;
    auto take = [&](size_t bytes) -> char* {
        char* p = ws; ws += (bytes + 255) & ~(size_t)255; return p;
    };
    float*    bufP = (float*)   take((size_t)N_NODES * FDIM * sizeof(float));
    float*    bufQ = (float*)   take((size_t)N_NODES * FDIM * sizeof(float));
    float*    Hbuf = (float*)   take((size_t)N_NODES * FDIM * sizeof(float));
    float*    als  = (float*)   take((size_t)N_NODES * NHEAD * sizeof(float));
    float*    ald  = (float*)   take((size_t)N_NODES * NHEAD * sizeof(float));
    unsigned* mbuf = (unsigned*)take((size_t)N_NODES * NHEAD * sizeof(unsigned));
    float*    ssum = (float*)   take((size_t)N_NODES * NHEAD * sizeof(float));
    float*    exb  = (float*)   take((size_t)EE * NHEAD * sizeof(float));
    float*    psum = (float*)   take((size_t)NGRAPH * FDIM * sizeof(float));
    float*    cnt  = (float*)   take((size_t)NGRAPH * sizeof(float));

    // skip branch: xs1 -> P, xs2 -> Q
    run_gat(x,    F_IN, Ws1, as1s, as1d, bs1, bufP, Hbuf, als, ald, mbuf, ssum, exb, ei, stream);
    run_gat(bufP, FDIM, Ws2, as2s, as2d, bs2, bufQ, Hbuf, als, ald, mbuf, ssum, exb, ei, stream);
    // main branch: h1 -> P (overwrites xs1, no longer needed), h2 -> P (in-place rotate)
    run_gat(x,    F_IN, W1,  a1s,  a1d,  b1,  bufP, Hbuf, als, ald, mbuf, ssum, exb, ei, stream);
    run_gat(bufP, FDIM, W2,  a2s,  a2d,  b2,  bufP, Hbuf, als, ald, mbuf, ssum, exb, ei, stream);

    // pool (h2 + xs2) per graph, then readout MLP
    const int gtot = NGRAPH * FDIM;
    zero_f_kernel<<<(gtot + 255) / 256, 256, 0, stream>>>(psum, gtot);
    zero_f_kernel<<<1, 256, 0, stream>>>(cnt, NGRAPH);
    const int tot = N_NODES * FDIM;
    pool_kernel<<<(tot + 255) / 256, 256, 0, stream>>>(bufP, bufQ, batch, psum, cnt);
    readout_kernel<<<NGRAPH, FDIM, 0, stream>>>(psum, cnt, r1w, r1b, r2w, r2b, out);
}
